// LocalVoxelEncoder_11381663334565
// MI455X (gfx1250) — matrix operand order repaired and checked by reading, not compiled, not run on hardware
//
#include <hip/hip_runtime.h>

// ---------------------------------------------------------------------------
// LocalVoxelEncoder for MI455X (gfx1250, wave32, WMMA + TDM).
//
// Reference collapses to:
//   out[b, c, f(k)*512 + f(i)] = (1/64) * sum_j relu( conv3d(x)[b,c,i,j,k] )
// and 0 everywhere else (the (i,k)->pixel map is injective, count==64).
//
// conv3d with 1 input channel == GEMM:  D(128ch x 64j) = W(128x27) x P(27x64)
// per (b,i,k).  K padded 27->32, computed with v_wmma_f32_16x16x32_f16.
// Weights are staged into LDS once per workgroup with the Tensor Data Mover
// (tensor_load_to_lds, waited via s_wait_tensorcnt).
// ---------------------------------------------------------------------------

typedef __attribute__((ext_vector_type(16))) _Float16 v16h;
typedef __attribute__((ext_vector_type(8)))  float    v8f;
typedef __attribute__((ext_vector_type(4)))  float    v4f;
typedef __attribute__((ext_vector_type(4)))  unsigned int v4u;
typedef __attribute__((ext_vector_type(4)))  int      v4i;
typedef __attribute__((ext_vector_type(8)))  int      v8i;

#define C_DIM  128
#define RESO   512
#define GRID   64
#define NTAPS  27
#define K_PAD  32
#define W_ELTS (C_DIM * NTAPS)   // 3456 f32 = 13824 bytes

// ---------------------------------------------------------------------------
// Kernel 1: zero the 256 MB output plane (bandwidth-dominant part of the op).
// ---------------------------------------------------------------------------
__global__ void zero_out_kernel(v4f* __restrict__ out, size_t n4) {
    size_t idx = (size_t)blockIdx.x * blockDim.x + threadIdx.x;
    if (idx < n4) {
        v4f z = {0.0f, 0.0f, 0.0f, 0.0f};
        __builtin_nontemporal_store(z, &out[idx]);
    }
}

// ---------------------------------------------------------------------------
// Kernel 2: per-(b,i,k) GEMM 128x64x27 via WMMA + ReLU + j-mean + scatter.
// Block = 256 threads (8 waves). Wave w owns channel tile [16w, 16w+16).
// ---------------------------------------------------------------------------
__global__ void __launch_bounds__(256)
conv_plane_kernel(const float* __restrict__ x,
                  const float* __restrict__ conv_w,
                  const float* __restrict__ conv_b,
                  float* __restrict__ out) {
    const int k = blockIdx.x;   // z-index of voxel column
    const int i = blockIdx.y;   // x-index of voxel column
    const int b = blockIdx.z;   // batch

    // Patch matrix, transposed: pT[j][kk] (N=64 rows of K_PAD=32 halfs).
    // Row = 64B, so a lane's B fragment (16 contiguous halfs) = 2x ds_load_b128.
    __shared__ __align__(128) _Float16 pT[GRID][K_PAD];
    // Weight matrix staged by the Tensor Data Mover (raw f32, 13824 B).
    __shared__ __align__(128) float wlds[W_ELTS];

    const int tid = threadIdx.x;

    // ---- TDM: DMA the 128x27 f32 weight matrix into LDS (wave 0 only; TDM
    //      ignores EXEC, branch is wave-uniform) ---------------------------
    if (tid < 32) {
        const unsigned long long ga = (unsigned long long)(uintptr_t)conv_w;
        const unsigned int lds_addr = (unsigned int)(uintptr_t)&wlds[0];

        // D# group 0: [1:0]=count=1, [63:32]=lds_addr, [120:64]=global_addr,
        //             [127:126]=type=2
        v4u g0;
        g0.x = 1u;
        g0.y = lds_addr;
        g0.z = (unsigned int)(ga & 0xffffffffu);
        g0.w = (unsigned int)((ga >> 32) & 0x1ffffffu) | (2u << 30);

        // D# group 1: 1-D transfer of W_ELTS 4-byte elements.
        //   [17:16]=data_size=2 (4B); [79:48]=tensor_dim0=W_ELTS;
        //   [111:80]=tensor_dim1=1; [127:112]=tile_dim0=W_ELTS;
        //   [207:160]=tensor_dim0_stride=W_ELTS; rest zero.
        v8i g1;
        g1[0] = (int)(2u << 16);
        g1[1] = (int)((W_ELTS & 0xffff) << 16);      // tensor_dim0 low 16
        g1[2] = (int)(((W_ELTS >> 16) & 0xffff) | (1u << 16)); // dim0 hi | dim1=1
        g1[3] = (int)((W_ELTS & 0xffff) << 16);      // tile_dim0
        g1[4] = 0;                                   // tile_dim1/2 unused
        g1[5] = (int)W_ELTS;                         // tensor_dim0_stride low 32
        g1[6] = 0;
        g1[7] = 0;

        v4i gz4 = {0, 0, 0, 0};                      // groups 2/3 unused (<=2D)
        v8i gz8 = {0, 0, 0, 0, 0, 0, 0, 0};
        __builtin_amdgcn_tensor_load_to_lds(g0, g1, gz4, gz4, gz8, 0);
        __builtin_amdgcn_s_wait_tensorcnt(0);
    }

    // ---- Stage P into LDS: 4 threads per j-row, 8 taps each -------------
    {
        const int j   = tid >> 2;
        const int kk0 = (tid & 3) * 8;
        const float* xb = x + (size_t)b * (GRID * GRID * GRID);
#pragma unroll
        for (int u = 0; u < 8; ++u) {
            const int kk = kk0 + u;
            float v = 0.0f;
            if (kk < NTAPS) {                       // taps 27..31 are zero pad
                const int kd = kk / 9;
                const int rm = kk - 9 * kd;
                const int kh = rm / 3;
                const int kw = rm - 3 * kh;
                const int ii = i + kd - 1;
                const int jj = j + kh - 1;
                const int k2 = k + kw - 1;
                if ((unsigned)ii < GRID && (unsigned)jj < GRID && (unsigned)k2 < GRID)
                    v = xb[((size_t)ii * GRID + jj) * GRID + k2];
            }
            pT[j][kk] = (_Float16)v;
        }
    }
    __syncthreads();   // pT + wlds both visible to all waves

    const int wave = tid >> 5;          // 0..7 : channel tile (M-tile)
    const int lane = tid & 31;
    const int g    = lane >> 4;         // half-wave group
    const int m    = lane & 15;
    const int c0   = wave * 16;         // first channel of this wave's tile

    // ---- A fragment: W(16x32) f16, contiguous K order (k = 16*g + h), same as B ------
    // lanes 0-15 : M=lane, halfs 0-15 -> K=0..15
    // lanes 16-31: M=lane-16, halfs 0-15 -> K=16..31
    v16h afrag;
    {
        const float* wc = &wlds[(c0 + m) * NTAPS];
#pragma unroll
        for (int h = 0; h < 16; ++h) {
            const int kk = g * 16 + h;
            afrag[h] = (kk < NTAPS) ? (_Float16)wc[kk] : (_Float16)0.0f;
        }
    }

    // Bias per C/D row r: channel = c0 + r + 8*g (C/D layout: VGPR r holds
    // M=r for lanes 0-15 and M=r+8 for lanes 16-31).
    float brow[8];
#pragma unroll
    for (int r = 0; r < 8; ++r) brow[r] = conv_b[c0 + r + 8 * g];

    // ---- 4 N-tiles of WMMA; bias+ReLU per element, accumulate over j ----
    float acc[8];
#pragma unroll
    for (int r = 0; r < 8; ++r) acc[r] = 0.0f;

#pragma unroll
    for (int nt = 0; nt < 4; ++nt) {
        const int n = nt * 16 + m;
        // B fragment: lane (N=n) needs K = g*16 + h, h=0..15 -> contiguous.
        const v16h bfrag = *(const v16h*)(&pT[n][g * 16]);
        v8f cz = {};
        v8f d = __builtin_amdgcn_wmma_f32_16x16x32_f16(
            /*neg_a=*/false, afrag, /*neg_b=*/false, bfrag,
            /*c_mod=*/(short)0, cz, /*reuse_a=*/false, /*reuse_b=*/false);
#pragma unroll
        for (int r = 0; r < 8; ++r) {
            const float v = d[r] + brow[r];
            acc[r] += fmaxf(v, 0.0f);               // ReLU before the j-mean
        }
    }

    // ---- Pixel index: injective (i,k) -> 512x512 bin ---------------------
    auto fmap = [](int idx) -> int {
        float t = (-0.5f + (float)idx * (1.0f / 63.0f)) / 1.101f + 0.5f;
        t = fminf(fmaxf(t, 0.0f), 1.0f - 1e-5f);
        return (int)(t * 512.0f);
    };
    const int p = fmap(i) + RESO * fmap(k);

    // ---- Reduce over the 16 lanes of each half-wave (sums the 64 j's
    //      together with the 4-N-tile accumulation) and store ---------------
#pragma unroll
    for (int r = 0; r < 8; ++r) {
        float v = acc[r];
#pragma unroll
        for (int off = 8; off >= 1; off >>= 1)
            v += __shfl_xor(v, off, 32);            // stays within 16-lane group
        if (m == 0) {
            const int c = c0 + r + 8 * g;
            out[((size_t)b * C_DIM + c) * (RESO * RESO) + p] = v * (1.0f / 64.0f);
        }
    }
}

// ---------------------------------------------------------------------------
extern "C" void kernel_launch(void* const* d_in, const int* in_sizes, int n_in,
                              void* d_out, int out_size, void* d_ws, size_t ws_size,
                              hipStream_t stream) {
    const float* x      = (const float*)d_in[0];   // (2,1,64,64,64) f32
    const float* conv_w = (const float*)d_in[1];   // (128,1,3,3,3)  f32
    const float* conv_b = (const float*)d_in[2];   // (128,)         f32
    float* out = (float*)d_out;                    // (2,128,512,512) f32

    // 1) zero the full plane output (most bins have count 0 -> mean 0)
    const size_t n4 = (size_t)out_size / 4;        // 16,777,216 float4 stores
    zero_out_kernel<<<(unsigned)((n4 + 255) / 256), 256, 0, stream>>>((v4f*)out, n4);

    // 2) conv + relu + j-mean + scatter: one block per (k, i, b)
    dim3 grid(GRID, GRID, 2);
    conv_plane_kernel<<<grid, 256, 0, stream>>>(x, conv_w, conv_b, out);
}
